// DeepAR_AS_23175643529626
// MI455X (gfx1250) — compile-verified, matrix operand order checked by reading, not run on hardware
//
#include <hip/hip_runtime.h>

// ---------------- model constants ----------------
#define B_     2048
#define SEQ_   168
#define W_     192
#define TGT_   7
#define HID_   128
#define GATES_ 512
#define INP_   65
#define K0P_   96     // padded layer-0 K (3 x 32)

typedef __attribute__((ext_vector_type(16))) __bf16 v16bf;
typedef __attribute__((ext_vector_type(8)))  float  v8f;

// ---------------- helpers ----------------
__device__ __forceinline__ float bf2f(unsigned short h) {
    unsigned u = ((unsigned)h) << 16;
    return __builtin_bit_cast(float, u);
}
__device__ __forceinline__ unsigned short f2bf(float f) {
    unsigned u = __builtin_bit_cast(unsigned, f);
    unsigned r = u + 0x7FFFu + ((u >> 16) & 1u);   // RNE
    return (unsigned short)(r >> 16);
}
__device__ __forceinline__ float sigmoidf_(float x) {
    return 1.0f / (1.0f + __expf(-x));
}

// WMMA 16-bit operand lane layout (ISA 7.12.2, 16x32 A-matrix):
// lane L: row = L%16, half g = L/16; dword j holds k-pair kbase(j)+8g,
// kbase(j) = j<4 ? 2j : 2j+8.  "Swizzled" storage puts each lane's 16 bf16
// contiguous: tile = [lane][16] (512 bf16), so a fetch is two b128 loads.
__device__ __forceinline__ v16bf load16sw(const unsigned short* p /*tile base + lane*16, 32B aligned*/) {
    union { v16bf v; uint4 q[2]; } r;
    const uint4* q = (const uint4*)p;
    r.q[0] = q[0];
    r.q[1] = q[1];
    return r.v;
}

// Gather a 16x32 operand from a row-major [rows][ldk] bf16 array (used for
// activation tiles staged row-major in the GEMM kernel).
__device__ __forceinline__ v16bf gather16(const unsigned short* base, int row, int ldk, int g) {
    union { v16bf v; unsigned u[8]; } r;
    const unsigned short* p = base + row * ldk + g * 8;
#pragma unroll
    for (int j = 0; j < 8; ++j) {
        int k = (j < 4) ? (2 * j) : (2 * j + 8);
        r.u[j] = *(const unsigned*)(p + k);
    }
    return r.v;
}

// Scatter position of element k (0..31 within a k-tile) for row m in a
// swizzled tile: returns bf16 index within the 512-element tile.
__device__ __forceinline__ int sw_index(int m, int khalf) {
    int ga  = (khalf >> 3) & 1;          // lane half
    int kk  = khalf & 7;
    int hi4 = (khalf >> 4) & 1;
    int j   = hi4 * 4 + (kk >> 1);
    int p   = kk & 1;
    int la  = (m & 15) + 16 * ga;
    return la * 16 + j * 2 + p;
}

// ---------------- kernel 0: pack weights into swizzled bf16 tiles ----------------
// dst layout: tiles ordered ntile-major then kt; tile = 256 dwords = [lane][8 dw].
__device__ __forceinline__ void pack_one(const float* __restrict__ W, int Ksrc, int Kpad,
                                         unsigned* __restrict__ dst, int i) {
    int T = i >> 8, rem = i & 255;
    int l = rem >> 3, j = rem & 7;
    int KT = Kpad >> 5;
    int ntile = T / KT, kt = T - ntile * KT;
    int n = ntile * 16 + (l & 15);
    int k = kt * 32 + ((j < 4) ? 2 * j : 2 * j + 8) + ((l >> 4) << 3);
    float v0 = (k     < Ksrc) ? W[n * Ksrc + k]     : 0.0f;
    float v1 = (k + 1 < Ksrc) ? W[n * Ksrc + k + 1] : 0.0f;
    dst[i] = (unsigned)f2bf(v0) | ((unsigned)f2bf(v1) << 16);
}

__global__ void pack_weights_kernel(const float* __restrict__ Wih0, const float* __restrict__ Whh0,
                                    const float* __restrict__ Wih1, const float* __restrict__ Whh1,
                                    unsigned* __restrict__ p0, unsigned* __restrict__ p1,
                                    unsigned* __restrict__ p2, unsigned* __restrict__ p3) {
    int i = blockIdx.x * blockDim.x + threadIdx.x;
    const int n0 = 32 * 3 * 256;   // Wih0: 32 ntiles x 3 kt x 256 dwords
    const int n1 = 32 * 4 * 256;   // K=128 weights: 32 x 4 x 256 dwords
    if (i < n0) { pack_one(Wih0, INP_, K0P_, p0, i); return; }
    i -= n0;
    if (i < n1) { pack_one(Whh0, HID_, HID_, p1, i); return; }
    i -= n1;
    if (i < n1) { pack_one(Wih1, HID_, HID_, p2, i); return; }
    i -= n1;
    if (i < n1) { pack_one(Whh1, HID_, HID_, p3, i); }
}

// ---------------- kernel 1: per-(b,ch) mean / std over W ----------------
__global__ void norm_stats_kernel(const float* __restrict__ x_enc,
                                  float* __restrict__ mean, float* __restrict__ stdv) {
    int i = blockIdx.x * blockDim.x + threadIdx.x;
    if (i >= B_ * TGT_) return;
    int b = i / TGT_, ch = i % TGT_;
    float s = 0.f, s2 = 0.f;
    for (int t = 0; t < W_; ++t) {
        float v = x_enc[(b * W_ + t) * TGT_ + ch];
        s += v; s2 += v * v;
    }
    float m = s * (1.0f / W_);
    float var = s2 * (1.0f / W_) - m * m;
    if (var < 0.f) var = 0.f;
    mean[i] = m;
    stdv[i] = sqrtf(var + 1e-5f);
}

// ---------------- kernel 2: assemble bf16 input features (B*W) x 96 ----------------
__global__ void build_inp_kernel(const float* __restrict__ given_enc, const float* __restrict__ x_enc,
                                 const float* __restrict__ x_mark,    const float* __restrict__ meta_x,
                                 const float* __restrict__ temb0, const float* __restrict__ temb1,
                                 const float* __restrict__ temb2, const float* __restrict__ memb0,
                                 const float* __restrict__ memb1,
                                 const float* __restrict__ mean, const float* __restrict__ stdv,
                                 unsigned short* __restrict__ inp) {
    int row = blockIdx.x;            // b*W + t
    int b = row / W_, t = row % W_;
    int f = threadIdx.x;             // 0..95
    float v = 0.0f;
    if (f < 7) {                                        // prev_y (normalized, shifted, clipped)
        int tt = t - 1; if (tt < 0) tt = 0; if (tt > SEQ_ - 1) tt = SEQ_ - 1;
        v = (x_enc[(b * W_ + tt) * TGT_ + f] - mean[b * TGT_ + f]) / stdv[b * TGT_ + f];
    } else if (f < 11) {                                // time numeric
        v = given_enc[(b * W_ + t) * 7 + (f - 7)];
    } else if (f < 35) {                                // time-cat embeddings 3 x 8
        int e = f - 11, which = e >> 3, sub = e & 7;
        int idx = (int)given_enc[(b * W_ + t) * 7 + 4 + which];
        const float* tab = (which == 0) ? temb0 : (which == 1) ? temb1 : temb2;
        v = tab[idx * 8 + sub];
    } else if (f < 39) {                                // covariates
        v = x_mark[(b * W_ + t) * 4 + (f - 35)];
    } else if (f < 41) {                                // meta numeric
        v = meta_x[b * 4 + (f - 39)];
    } else if (f < 57) {                                // meta emb 0 (16)
        int idx = (int)meta_x[b * 4 + 2];
        v = memb0[idx * 16 + (f - 41)];
    } else if (f < 65) {                                // meta emb 1 (8)
        int idx = (int)meta_x[b * 4 + 3];
        v = memb1[idx * 8 + (f - 57)];
    }                                                   // 65..95 zero pad
    inp[(size_t)row * K0P_ + f] = f2bf(v);
}

// ---------------- kernel 3: WMMA GEMM  xp = A(K) . W^T + bih + bhh ----------------
// M-tile = (batch-tile of 16, one t). Wave w handles t = t0 + w, 4 N-tiles.
// Output stored in recurrence-native layout:
//   xpsw[ ((btile*W + t)*32 + ntile)*256 + lane*8 + r ]  (bf16)
// so the recurrence reads one b128 per gate tile per lane.
__global__ __launch_bounds__(256) void gemm_xp_kernel(const unsigned short* __restrict__ A, int K,
                                                      const unsigned short* __restrict__ Wsw,
                                                      const float* __restrict__ bih,
                                                      const float* __restrict__ bhh,
                                                      unsigned short* __restrict__ xpsw) {
    __shared__ __align__(16) unsigned short sA[8 * 16 * 128];   // [w][m][K], 32 KB max
    __shared__ __align__(16) unsigned short sB[4 * 4 * 512];    // 4 ntiles x KT swizzled tiles, 16 KB max
    int tid   = threadIdx.x;
    int btile = blockIdx.x / 24;         // W_/8 = 24 t-chunks
    int tch   = blockIdx.x % 24;
    int b0    = btile * 16;
    int t0    = tch * 8;
    int n0t   = blockIdx.y * 4;          // first ntile (of 32)
    int KT    = K >> 5;

    {   // stage A rows: logical row rr = w*16 + m  ->  global row (b0+m)*W + t0 + w
        int kd = K >> 1;                 // dwords per row
        unsigned* lA = (unsigned*)sA;
        for (int i = tid; i < 128 * kd; i += 256) {
            int rr = i / kd, c = i - rr * kd;
            int w = rr >> 4, m = rr & 15;
            const unsigned* src = (const unsigned*)(A + (size_t)((b0 + m) * W_ + t0 + w) * K);
            lA[i] = src[c];
        }
        // stage B: contiguous span of 4*KT swizzled tiles
        const unsigned* gB = (const unsigned*)Wsw + (size_t)n0t * KT * 256;
        unsigned* lB = (unsigned*)sB;
        for (int i = tid; i < 4 * KT * 256; i += 256) lB[i] = gB[i];
    }
    __syncthreads();

    int w = tid >> 5, lane = tid & 31;
    int lrow = lane & 15, g = lane >> 4;

    v8f acc[4];
#pragma unroll
    for (int nt = 0; nt < 4; ++nt) acc[nt] = (v8f){0, 0, 0, 0, 0, 0, 0, 0};

    for (int kt = 0; kt < KT; ++kt) {
        v16bf a = gather16(sA + (w * 16) * K + kt * 32, lrow, K, g);
#pragma unroll
        for (int nt = 0; nt < 4; ++nt) {
            v16bf bmat = load16sw(sB + (nt * KT + kt) * 512 + lane * 16);
            acc[nt] = __builtin_amdgcn_wmma_f32_16x16x32_bf16(
                false, a, false, bmat, (short)0, acc[nt], false, false);
        }
    }

    int t = t0 + w;
#pragma unroll
    for (int nt = 0; nt < 4; ++nt) {
        int n = (n0t + nt) * 16 + lrow;
        float bsum = bih[n] + bhh[n];
        union { uint4 q; unsigned short s[8]; } pk;
#pragma unroll
        for (int r = 0; r < 8; ++r) pk.s[r] = f2bf(acc[nt][r] + bsum);
        *(uint4*)(xpsw + ((size_t)(btile * W_ + t) * 32 + (n0t + nt)) * 256 + lane * 8) = pk.q;
    }
}

// ---------------- kernel 4: persistent LSTM recurrence (WMMA) ----------------
// 128 WGs x 16 batch rows, 192 steps. Wave w owns hid columns [w*16, w*16+16)
// of each gate block -> i/f/g/o for one (row,hid) land in the same lane; cell
// state lives in registers. Whh cached in LDS pre-swizzled (b128 fetches);
// h(t) ping-pongs between two A-swizzled LDS tile sets -> ONE barrier/step.
// xp (pre-activations) are software-pipelined one step ahead.
__global__ __launch_bounds__(256) void lstm_rec_kernel(const unsigned short* __restrict__ xpsw,
                                                       const unsigned short* __restrict__ WhhSw,
                                                       unsigned short* __restrict__ hout,
                                                       const float* __restrict__ Wm, const float* __restrict__ bm,
                                                       const float* __restrict__ Ws, const float* __restrict__ bs,
                                                       const float* __restrict__ mean, const float* __restrict__ stdv,
                                                       float* __restrict__ out, int layer1) {
    __shared__ __align__(16) unsigned short sWhh[GATES_ * HID_];   // 128 KB, swizzled tiles
    __shared__ __align__(16) unsigned short sHsw[2][4 * 512];      // 8 KB, ping-pong h tiles
    __shared__ __align__(16) float sWo[14 * HID_];                 // 7 KB
    __shared__ float sBo[14];

    int tid   = threadIdx.x;
    int btile = blockIdx.x;
    int b0    = btile * 16;

    {   // cache swizzled Whh in LDS (persistent for 192 steps)
        const uint4* gw = (const uint4*)WhhSw;
        uint4* lw = (uint4*)sWhh;
        for (int i = tid; i < GATES_ * HID_ * 2 / 16; i += 256) lw[i] = gw[i];
    }
    for (int i = tid; i < 2 * 4 * 512 / 2; i += 256) ((unsigned*)sHsw)[i] = 0u;   // h0 = 0
    if (layer1) {
        for (int i = tid; i < TGT_ * HID_; i += 256) { sWo[i] = Wm[i]; sWo[TGT_ * HID_ + i] = Ws[i]; }
        if (tid < TGT_) { sBo[tid] = bm[tid]; sBo[TGT_ + tid] = bs[tid]; }
    }

    float cst[8];
#pragma unroll
    for (int r = 0; r < 8; ++r) cst[r] = 0.0f;

    int w = tid >> 5, lane = tid & 31;
    int lrow = lane & 15, gg = lane >> 4;
    int nbase = w * 16 + lrow;                 // hidden-unit column this lane owns
    int ktw = nbase >> 5, khw = nbase & 31;    // scatter coords for h writes

    // xp base for this (btile, wave, lane); advances by 32*256 bf16 per step
    const unsigned short* xrow = xpsw + (size_t)(btile * W_) * 32 * 256 + (size_t)w * 256 + lane * 8;

    union U4 { uint4 q; unsigned short s[8]; };
    U4 xq[4];
#pragma unroll
    for (int g4 = 0; g4 < 4; ++g4)             // preload xp(t=0)
        xq[g4].q = *(const uint4*)(xrow + (size_t)g4 * 8 * 256);

    __syncthreads();

    for (int t = 0; t < W_; ++t) {
        const unsigned short* rbuf = sHsw[t & 1];        // h(t-1)
        unsigned short*       wbuf = sHsw[(t + 1) & 1];  // h(t)

        // C init from pipelined xp registers
        v8f acc[4];
#pragma unroll
        for (int g4 = 0; g4 < 4; ++g4)
#pragma unroll
            for (int r = 0; r < 8; ++r) acc[g4][r] = bf2f(xq[g4].s[r]);

        // prefetch xp(t+1) -- overlaps the WMMA/DS work below
        if (t + 1 < W_) {
            const unsigned short* xnext = xrow + (size_t)(t + 1) * 32 * 256;
#pragma unroll
            for (int g4 = 0; g4 < 4; ++g4)
                xq[g4].q = *(const uint4*)(xnext + (size_t)g4 * 8 * 256);
        }

        // h(t-1) @ Whh^T : kt-outer / gate-inner -> shared A per group,
        // accumulator RAW distance 4.
#pragma unroll
        for (int kt = 0; kt < 4; ++kt) {
            v16bf a = load16sw(rbuf + kt * 512 + lane * 16);
#pragma unroll
            for (int g4 = 0; g4 < 4; ++g4) {
                v16bf bmat = load16sw(sWhh + ((g4 * 8 + w) * 4 + kt) * 512 + lane * 16);
                acc[g4] = __builtin_amdgcn_wmma_f32_16x16x32_bf16(
                    false, a, false, bmat, (short)0, acc[g4], false, false);
            }
        }

        // LSTM cell update: lane holds i/f/g/o for rows m=r+8*gg, hid=nbase.
        // Writes go to the OTHER buffer -> no pre-write barrier needed.
#pragma unroll
        for (int r = 0; r < 8; ++r) {
            float iv = sigmoidf_(acc[0][r]);
            float fv = sigmoidf_(acc[1][r]);
            float gv = tanhf(acc[2][r]);
            float ov = sigmoidf_(acc[3][r]);
            float c  = fv * cst[r] + iv * gv;
            cst[r] = c;
            float h = ov * tanhf(c);
            int m = r + 8 * gg;
            unsigned short hb = f2bf(h);
            wbuf[ktw * 512 + sw_index(m, khw)] = hb;               // A-swizzled scatter
            if (hout) hout[(size_t)((b0 + m) * W_ + t) * HID_ + nbase] = hb;  // row-major for next GEMM
        }
        __syncthreads();   // h(t) visible; next iter reads wbuf, writes rbuf

        if (layer1 && tid < 16 * 14) {         // fused mean/sigma head + de-norm
            int m = tid / 14, oi = tid % 14;
            const float* wrow = sWo + oi * HID_;
            float s = 0.0f;
#pragma unroll 16
            for (int k = 0; k < HID_; ++k) {
                int idx = (k >> 5) * 512 + sw_index(m, k & 31);
                s += bf2f(wbuf[idx]) * wrow[k];
            }
            s += sBo[oi];
            int bi = b0 + m;
            float ov;
            if (oi < 7) {
                ov = s * stdv[bi * TGT_ + oi] + mean[bi * TGT_ + oi];
            } else {
                float sp = (s > 20.0f) ? s : log1pf(__expf(s));
                ov = sp * stdv[bi * TGT_ + (oi - 7)];
            }
            out[(size_t)(bi * W_ + t) * 14 + oi] = ov;
        }
        // proj reads wbuf; next iter writes rbuf -> no conflict, no extra barrier.
    }
}

// ---------------- workspace layout (bytes) ----------------
static const size_t OFF_WIH0P = 0;                                     //  98304
static const size_t OFF_WHH0P = OFF_WIH0P + (size_t)GATES_ * K0P_ * 2; // +131072
static const size_t OFF_WIH1P = OFF_WHH0P + (size_t)GATES_ * HID_ * 2;
static const size_t OFF_WHH1P = OFF_WIH1P + (size_t)GATES_ * HID_ * 2;
static const size_t OFF_MEAN  = OFF_WHH1P + (size_t)GATES_ * HID_ * 2;
static const size_t OFF_STD   = OFF_MEAN + (size_t)B_ * TGT_ * 4;
static const size_t OFF_INP   = OFF_STD + (size_t)B_ * TGT_ * 4;
static const size_t OFF_H1    = OFF_INP + (size_t)B_ * W_ * K0P_ * 2;   // ~75.5 MB
static const size_t OFF_XP    = OFF_H1 + (size_t)B_ * W_ * HID_ * 2;    // ~100.7 MB
// OFF_XP span: B*W*512*2 = ~402.7 MB ; total ws ~ 580 MB

extern "C" void kernel_launch(void* const* d_in, const int* in_sizes, int n_in,
                              void* d_out, int out_size, void* d_ws, size_t ws_size,
                              hipStream_t stream) {
    const float* given_enc = (const float*)d_in[0];
    const float* x_enc     = (const float*)d_in[1];
    const float* x_mark    = (const float*)d_in[2];
    const float* meta_x    = (const float*)d_in[3];
    const float* temb0     = (const float*)d_in[4];
    const float* temb1     = (const float*)d_in[5];
    const float* temb2     = (const float*)d_in[6];
    const float* memb0     = (const float*)d_in[7];
    const float* memb1     = (const float*)d_in[8];
    const float* Wih0      = (const float*)d_in[9];
    const float* Whh0      = (const float*)d_in[10];
    const float* bih0      = (const float*)d_in[11];
    const float* bhh0      = (const float*)d_in[12];
    const float* Wih1      = (const float*)d_in[13];
    const float* Whh1      = (const float*)d_in[14];
    const float* bih1      = (const float*)d_in[15];
    const float* bhh1      = (const float*)d_in[16];
    const float* Wm        = (const float*)d_in[17];
    const float* bm        = (const float*)d_in[18];
    const float* Ws        = (const float*)d_in[19];
    const float* bs        = (const float*)d_in[20];

    char* ws = (char*)d_ws;
    unsigned short* wih0p = (unsigned short*)(ws + OFF_WIH0P);
    unsigned short* whh0p = (unsigned short*)(ws + OFF_WHH0P);
    unsigned short* wih1p = (unsigned short*)(ws + OFF_WIH1P);
    unsigned short* whh1p = (unsigned short*)(ws + OFF_WHH1P);
    float*          meanp = (float*)(ws + OFF_MEAN);
    float*          stdp  = (float*)(ws + OFF_STD);
    unsigned short* inp   = (unsigned short*)(ws + OFF_INP);
    unsigned short* h1    = (unsigned short*)(ws + OFF_H1);
    unsigned short* xpb   = (unsigned short*)(ws + OFF_XP);
    float* outp = (float*)d_out;

    // 0) weight packing -> swizzled bf16 WMMA tiles
    {
        int total = 32 * 3 * 256 + 3 * (32 * 4 * 256);   // dwords
        pack_weights_kernel<<<(total + 255) / 256, 256, 0, stream>>>(
            Wih0, Whh0, Wih1, Whh1,
            (unsigned*)wih0p, (unsigned*)whh0p, (unsigned*)wih1p, (unsigned*)whh1p);
    }
    // 1) normalization stats
    norm_stats_kernel<<<(B_ * TGT_ + 255) / 256, 256, 0, stream>>>(x_enc, meanp, stdp);
    // 2) input feature assembly
    build_inp_kernel<<<B_ * W_, K0P_, 0, stream>>>(
        given_enc, x_enc, x_mark, meta_x, temb0, temb1, temb2, memb0, memb1,
        meanp, stdp, inp);
    // 3) xp0 = inp . Wih0^T + b   (K = 96)
    dim3 gg(128 * 24, GATES_ / 64);
    gemm_xp_kernel<<<gg, 256, 0, stream>>>(inp, K0P_, wih0p, bih0, bhh0, xpb);
    // 4) LSTM layer 0 -> h1
    lstm_rec_kernel<<<B_ / 16, 256, 0, stream>>>(
        xpb, whh0p, h1, Wm, bm, Ws, bs, meanp, stdp, outp, 0);
    // 5) xp1 = h1 . Wih1^T + b   (K = 128), reuse xp slab
    gemm_xp_kernel<<<gg, 256, 0, stream>>>(h1, HID_, wih1p, bih1, bhh1, xpb);
    // 6) LSTM layer 1 + fused mean/sigma head -> d_out
    lstm_rec_kernel<<<B_ / 16, 256, 0, stream>>>(
        xpb, whh1p, nullptr, Wm, bm, Ws, bs, meanp, stdp, outp, 1);
}
